// Attention_67388036874372
// MI455X (gfx1250) — matrix operand order repaired and checked by reading, not compiled, not run on hardware
//
#include <hip/hip_runtime.h>
#include <hip/hip_bf16.h>
#include <math.h>

#define H 512
#define S 65536
#define ROWS 32            // S-rows per block in main kernel (2 M-tiles)
#define MT 2               // M-tiles per wave
#define BLOCK 256          // 8 waves
#define WAVES 8
#define NT_PER_WAVE 4      // 32 n-tiles / 8 waves
#define KT (H / 32)        // 16 k-tiles of 32
#define NTILES (H / 16)    // 32
#define CTX_BLOCKS 64

typedef __attribute__((ext_vector_type(16))) __bf16 v16bf;
typedef __attribute__((ext_vector_type(8)))  float  v8f;

__device__ __forceinline__ unsigned short f2bf(float f) {
    unsigned u = __float_as_uint(f);
    u += 0x7FFFu + ((u >> 16) & 1u);   // round-to-nearest-even
    return (unsigned short)(u >> 16);
}

// ---------------- tdec[o] = sum_h dec[h] * Wdec[o,h] ----------------
__global__ void k_tdec(const float* __restrict__ dec,
                       const float* __restrict__ Wdec,
                       float* __restrict__ tdec) {
    int o = threadIdx.x;
    const float* row = Wdec + o * H;
    float acc = 0.f;
    for (int h = 0; h < H; ++h) acc += dec[h] * row[h];
    tdec[o] = acc;
}

// ------- pack Wenc^T into per-lane bf16 B-fragments (16x16x32 layout) -------
// B[k][n] = Wenc[n][k].  Fragment (nt,kt): lane = n + 16*halfK,
// element i of v16bf = K = kt*32 + halfK*16 + i  (pairs pack into VGPRs).
__global__ void k_packW(const float* __restrict__ Wenc,
                        unsigned short* __restrict__ wp) {
    int frag = blockIdx.x;          // nt*KT + kt, 512 frags
    int lane = threadIdx.x;         // 0..31
    int nt = frag / KT, kt = frag % KT;
    int n = lane & 15, halfK = lane >> 4;
    const float* src = Wenc + (size_t)(nt * 16 + n) * H + kt * 32 + halfK * 16;
    unsigned short* dst = wp + ((size_t)frag * 32 + lane) * 16;
    #pragma unroll
    for (int i = 0; i < 16; ++i) dst[i] = f2bf(src[i]);
}

// ---------------- main: temp_enc GEMM + tanh + va-dot + mask ----------------
__global__ void __launch_bounds__(BLOCK) k_main(
    const float* __restrict__ enc, const unsigned short* __restrict__ wp,
    const float* __restrict__ tdec, const float* __restrict__ va,
    const unsigned char* __restrict__ mask, float* __restrict__ energies) {

    __shared__ __align__(16) unsigned short tile[ROWS * 520]; // 520 = 512 + pad(8)
    __shared__ float e_part[WAVES][ROWS];

    const int tid  = threadIdx.x;
    const int lane = tid & 31;
    const int wave = tid >> 5;
    const int sbase = blockIdx.x * ROWS;

    // ---- stage 32x512 enc rows as bf16 in LDS (coalesced float4 loads) ----
    const float4* g = reinterpret_cast<const float4*>(enc + (size_t)sbase * H);
    #pragma unroll
    for (int it = 0; it < (ROWS * H / 4) / BLOCK; ++it) {      // 16 iters
        int idx4 = tid + it * BLOCK;
        float4 f = g[idx4];
        int row = idx4 >> 7;                                   // 128 float4/row
        int col4 = idx4 & 127;
        uint2 p;
        p.x = (unsigned)f2bf(f.x) | ((unsigned)f2bf(f.y) << 16);
        p.y = (unsigned)f2bf(f.z) | ((unsigned)f2bf(f.w) << 16);
        *reinterpret_cast<uint2*>(&tile[row * 520 + col4 * 4]) = p;
    }
    __syncthreads();

    const int halfK = lane >> 4;
    const int mrow  = lane & 15;
    const unsigned short* arow0 = &tile[mrow * 520];           // M-tile 0
    const unsigned short* arow1 = &tile[(mrow + 16) * 520];    // M-tile 1

    // Uniform base for this wave's 4 n-tiles; lane offset is loop-invariant;
    // (nt*KT + kt)*512 shorts folds into the load's immediate offset.
    const unsigned short* __restrict__ wpw =
        wp + (size_t)wave * (NT_PER_WAVE * KT * 32 * 16) + lane * 16;

    v8f c[MT][NT_PER_WAVE] = {};
    #pragma unroll 4
    for (int kt = 0; kt < KT; ++kt) {
        union { v16bf v; uint4 q[2]; } A0, A1;
        A0.q[0] = *reinterpret_cast<const uint4*>(arow0 + kt * 32 + halfK * 16);
        A0.q[1] = *reinterpret_cast<const uint4*>(arow0 + kt * 32 + halfK * 16 + 8);
        A1.q[0] = *reinterpret_cast<const uint4*>(arow1 + kt * 32 + halfK * 16);
        A1.q[1] = *reinterpret_cast<const uint4*>(arow1 + kt * 32 + halfK * 16 + 8);
        #pragma unroll
        for (int nt = 0; nt < NT_PER_WAVE; ++nt) {
            const v16bf b = *reinterpret_cast<const v16bf*>(
                wpw + (nt * KT + kt) * 512);
            c[0][nt] = __builtin_amdgcn_wmma_f32_16x16x32_bf16(
                false, A0.v, false, b, (short)0, c[0][nt], false, false);
            c[1][nt] = __builtin_amdgcn_wmma_f32_16x16x32_bf16(
                false, A1.v, false, b, (short)0, c[1][nt], false, false);
        }
    }

    // ---- epilogue: x = tanh(tdec + temp_enc); e += va*x ----
    // C layout: VGPR j -> M = j + 8*halfK (+16*mt), N = lane&15
    float esum[MT][8];
    #pragma unroll
    for (int mt = 0; mt < MT; ++mt)
        #pragma unroll
        for (int j = 0; j < 8; ++j) esum[mt][j] = 0.f;
    #pragma unroll
    for (int nt = 0; nt < NT_PER_WAVE; ++nt) {
        int n = (wave * NT_PER_WAVE + nt) * 16 + mrow;
        float td = tdec[n];
        float vn = va[n];
        #pragma unroll
        for (int mt = 0; mt < MT; ++mt)
            #pragma unroll
            for (int j = 0; j < 8; ++j)
                esum[mt][j] += vn * tanhf(td + c[mt][nt][j]);
    }
    // deterministic reduction over the 16 lanes of each half (N dimension)
    #pragma unroll
    for (int m = 1; m <= 8; m <<= 1)
        #pragma unroll
        for (int mt = 0; mt < MT; ++mt)
            #pragma unroll
            for (int j = 0; j < 8; ++j)
                esum[mt][j] += __shfl_xor(esum[mt][j], m, 32);
    if ((lane & 15) == 0) {
        #pragma unroll
        for (int mt = 0; mt < MT; ++mt)
            #pragma unroll
            for (int j = 0; j < 8; ++j)
                e_part[wave][mt * 16 + j + 8 * halfK] = esum[mt][j];
    }
    __syncthreads();
    if (tid < ROWS) {
        float e = 0.f;
        #pragma unroll
        for (int w = 0; w < WAVES; ++w) e += e_part[w][tid];
        int s = sbase + tid;
        energies[s] = mask[s] ? e : -INFINITY;
    }
}

// ---------------- softmax stats: max and 1/sum(exp) ----------------
__global__ void __launch_bounds__(1024) k_reduce(const float* __restrict__ e,
                                                 float* __restrict__ stats) {
    __shared__ float red[1024];
    int tid = threadIdx.x;
    float m = -INFINITY;
    for (int i = tid; i < S; i += 1024) m = fmaxf(m, e[i]);
    red[tid] = m; __syncthreads();
    for (int off = 512; off > 0; off >>= 1) {
        if (tid < off) red[tid] = fmaxf(red[tid], red[tid + off]);
        __syncthreads();
    }
    float gmax = red[0];
    __syncthreads();
    float s = 0.f;
    for (int i = tid; i < S; i += 1024) s += expf(e[i] - gmax);
    red[tid] = s; __syncthreads();
    for (int off = 512; off > 0; off >>= 1) {
        if (tid < off) red[tid] += red[tid + off];
        __syncthreads();
    }
    if (tid == 0) { stats[0] = gmax; stats[1] = 1.0f / red[0]; }
}

// -------- weights (in-place e->w) + per-block context partials --------
__global__ void __launch_bounds__(256) k_ctx(const float* __restrict__ enc,
                                             float* __restrict__ weights,
                                             const float* __restrict__ stats,
                                             float* __restrict__ partial) {
    __shared__ float wbuf[256];
    int tid = threadIdx.x;
    int blk = blockIdx.x;
    float gmax = stats[0], inv = stats[1];
    int h0 = tid * 2;
    float2 acc = {0.f, 0.f};
    for (int it = 0; it < (S / CTX_BLOCKS) / 256; ++it) {      // 4 iters
        int r0 = blk * (S / CTX_BLOCKS) + it * 256;
        float w = expf(weights[r0 + tid] - gmax) * inv;
        weights[r0 + tid] = w;
        wbuf[tid] = w;
        __syncthreads();
        for (int r = 0; r < 256; ++r) {
            float ws = wbuf[r];
            float2 ev = *reinterpret_cast<const float2*>(
                enc + (size_t)(r0 + r) * H + h0);
            acc.x += ws * ev.x;
            acc.y += ws * ev.y;
        }
        __syncthreads();
    }
    *reinterpret_cast<float2*>(partial + (size_t)blk * H + h0) = acc;
}

__global__ void k_final(const float* __restrict__ partial, float* __restrict__ ctx) {
    int h = threadIdx.x;
    float s = 0.f;
    for (int b = 0; b < CTX_BLOCKS; ++b) s += partial[(size_t)b * H + h];
    ctx[h] = s;
}

extern "C" void kernel_launch(void* const* d_in, const int* in_sizes, int n_in,
                              void* d_out, int out_size, void* d_ws, size_t ws_size,
                              hipStream_t stream) {
    (void)in_sizes; (void)n_in; (void)out_size; (void)ws_size;
    const float* enc  = (const float*)d_in[0];           // [1,S,H]
    const float* dec  = (const float*)d_in[1];           // [1,H]
    const unsigned char* mask = (const unsigned char*)d_in[2]; // [S] bool
    const float* Wenc = (const float*)d_in[3];           // [H,H]
    const float* Wdec = (const float*)d_in[4];           // [H,H]
    const float* va   = (const float*)d_in[5];           // [1,H]

    float* out = (float*)d_out;
    float* ctx = out;                 // [H]   context
    float* weights = out + H;         // [S]   attention weights (also energy scratch)

    char* ws = (char*)d_ws;
    float* tdec    = (float*)(ws);                              // 2 KB
    float* stats   = (float*)(ws + 2048);                       // 8 B (padded)
    float* partial = (float*)(ws + 4096);                       // 128 KB
    unsigned short* wp =
        (unsigned short*)(ws + 4096 + CTX_BLOCKS * H * sizeof(float)); // 512 KB

    k_tdec  <<<1, H, 0, stream>>>(dec, Wdec, tdec);
    k_packW <<<NTILES * KT, 32, 0, stream>>>(Wenc, wp);
    k_main  <<<S / ROWS, BLOCK, 0, stream>>>(enc, wp, tdec, va, mask, weights);
    k_reduce<<<1, 1024, 0, stream>>>(weights, stats);
    k_ctx   <<<CTX_BLOCKS, 256, 0, stream>>>(enc, weights, stats, partial);
    k_final <<<1, H, 0, stream>>>(partial, ctx);
}
